// GraphNN_45603962748997
// MI455X (gfx1250) — compile-verified, matrix operand order checked by reading, not run on hardware
//
#include <hip/hip_runtime.h>
#include <hip/hip_bf16.h>

#define NN 100000
#define NE 1000000
#define ND 128
#define ED 32
#define HID 64
#define HIER 32
#define NCLS 10
#define BN_EPS 1e-5f

typedef __attribute__((ext_vector_type(16))) _Float16 v16h;
typedef __attribute__((ext_vector_type(8)))  float    v8f;

__device__ __forceinline__ void atomAddF(float* p, float v) {
    unsafeAtomicAdd(p, v);
}

// ---------------------------------------------------------------------------
// Weight packing: source f32 W[K][Ncol] row-major -> f16 fragments laid out
// [ktile][ctile][lane][16] so each wave B-fragment is one contiguous 32B load.
// 16-bit B 32x16 layout: lane<16 -> col=lane,    K in {0..7}U{16..23}
//                        lane>=16 -> col=lane-16, K in {8..15}U{24..31}
// ---------------------------------------------------------------------------
__global__ void pack_w(const float* __restrict__ W, _Float16* __restrict__ P,
                       int K, int Ncol, int NcolPad) {
    int idx = blockIdx.x * blockDim.x + threadIdx.x;
    int ktiles = K / 32, ctiles = NcolPad / 16;
    int total = ktiles * ctiles * 32;
    if (idx >= total) return;
    int lane = idx & 31;
    int tile = idx >> 5;
    int ct = tile % ctiles, kt = tile / ctiles;
    int c  = ct * 16 + (lane & 15);
    int kh = (lane >> 4) * 8;
    _Float16* o = P + (size_t)idx * 16;
#pragma unroll
    for (int j = 0; j < 8; ++j) {
        int k0 = kt * 32 + kh + j;
        int k1 = kt * 32 + 16 + kh + j;
        float v0 = (c < Ncol) ? W[(size_t)k0 * Ncol + c] : 0.f;
        float v1 = (c < Ncol) ? W[(size_t)k1 * Ncol + c] : 0.f;
        o[j]     = (_Float16)v0;
        o[8 + j] = (_Float16)v1;
    }
}

// A fragment (16x32 f16) from f32 row-major source.
// lane<16: row=lane, K in {0..7}U{16..23}; lane>=16: row=lane-16, K {8..15}U{24..31}
__device__ __forceinline__ v16h load_a_f32(const float* __restrict__ A,
                                           int row0, int ldA, int k0) {
    int lane = threadIdx.x & 31;
    const float* p = A + (size_t)(row0 + (lane & 15)) * ldA + k0 + ((lane >> 4) * 8);
    v16h a;
#pragma unroll
    for (int j = 0; j < 8; ++j) {
        a[j]     = (_Float16)p[j];
        a[8 + j] = (_Float16)p[16 + j];
    }
    return a;
}

// ---------------------------------------------------------------------------
// Generic GEMM: Out[M, ncols] = A[M, 32*KTILES] @ Wpacked + bias
// One wave per 16-row tile, NTILES 16-col accumulators, f32 accumulate.
// ---------------------------------------------------------------------------
template <int KTILES, int NTILES>
__global__ void gemm_wmma(const float* __restrict__ A, int ldA,
                          const _Float16* __restrict__ Wp,
                          const float* __restrict__ bias,
                          float* __restrict__ Out, int ldOut, int ncols, int mtiles) {
    int wave = blockIdx.x * (blockDim.x >> 5) + (threadIdx.x >> 5);
    if (wave >= mtiles) return;
    int row0 = wave * 16;
    int lane = threadIdx.x & 31;
    int coln = lane & 15;
    int mh   = (lane >> 4) * 8;

    v8f acc[NTILES];
#pragma unroll
    for (int t = 0; t < NTILES; ++t) {
        int col = t * 16 + coln;
        float b = (col < ncols) ? bias[col] : 0.f;
#pragma unroll
        for (int r = 0; r < 8; ++r) acc[t][r] = b;
    }
#pragma unroll
    for (int kt = 0; kt < KTILES; ++kt) {
        v16h a = load_a_f32(A, row0, ldA, kt * 32);
#pragma unroll
        for (int t = 0; t < NTILES; ++t) {
            v16h b = *(const v16h*)(Wp + ((size_t)(kt * NTILES + t) * 32 + lane) * 16);
            acc[t] = __builtin_amdgcn_wmma_f32_16x16x32_f16(
                false, a, false, b, (short)0, acc[t], false, false);
        }
    }
#pragma unroll
    for (int t = 0; t < NTILES; ++t) {
        int col = t * 16 + coln;
        if (col < ncols) {
#pragma unroll
            for (int r = 0; r < 8; ++r)
                Out[(size_t)(row0 + mh + r) * ldOut + col] = acc[t][r];
        }
    }
}

// Concat GEMM: A = [h_sum (64) | c (32)], K=96 -> 3 ktiles
__global__ void gemm_out1(const float* __restrict__ hsum, const float* __restrict__ c,
                          const _Float16* __restrict__ Wp, const float* __restrict__ bias,
                          float* __restrict__ Out, int mtiles) {
    int wave = blockIdx.x * (blockDim.x >> 5) + (threadIdx.x >> 5);
    if (wave >= mtiles) return;
    int row0 = wave * 16;
    int lane = threadIdx.x & 31;
    int coln = lane & 15;
    int mh   = (lane >> 4) * 8;

    v8f acc[8];
#pragma unroll
    for (int t = 0; t < 8; ++t) {
        float b = bias[t * 16 + coln];
#pragma unroll
        for (int r = 0; r < 8; ++r) acc[t][r] = b;
    }
#pragma unroll
    for (int kt = 0; kt < 3; ++kt) {
        v16h a = (kt < 2) ? load_a_f32(hsum, row0, 64, kt * 32)
                          : load_a_f32(c,    row0, 32, 0);
#pragma unroll
        for (int t = 0; t < 8; ++t) {
            v16h b = *(const v16h*)(Wp + ((size_t)(kt * 8 + t) * 32 + lane) * 16);
            acc[t] = __builtin_amdgcn_wmma_f32_16x16x32_f16(
                false, a, false, b, (short)0, acc[t], false, false);
        }
    }
#pragma unroll
    for (int t = 0; t < 8; ++t)
#pragma unroll
        for (int r = 0; r < 8; ++r)
            Out[(size_t)(row0 + mh + r) * 128 + t * 16 + coln] = acc[t][r];
}

// Final GEMM with fused BatchNorm (scale/shift) + PReLU in the A loader.
__global__ void gemm_out2(const float* __restrict__ z, const float* __restrict__ scsh,
                          const float* __restrict__ prelu,
                          const _Float16* __restrict__ Wp, const float* __restrict__ bias,
                          float* __restrict__ Out, int mtiles) {
    int wave = blockIdx.x * (blockDim.x >> 5) + (threadIdx.x >> 5);
    if (wave >= mtiles) return;
    int row0 = wave * 16;
    int lane = threadIdx.x & 31;
    int coln = lane & 15;
    int mh   = (lane >> 4) * 8;
    float pa = prelu[0];

    v8f acc;
    {
        float b = (coln < NCLS) ? bias[coln] : 0.f;
#pragma unroll
        for (int r = 0; r < 8; ++r) acc[r] = b;
    }
#pragma unroll
    for (int kt = 0; kt < 4; ++kt) {
        int kh = (lane >> 4) * 8;
        const float* p = z + (size_t)(row0 + (lane & 15)) * 128 + kt * 32 + kh;
        int kb = kt * 32 + kh;
        v16h a;
#pragma unroll
        for (int j = 0; j < 8; ++j) {
            int k0 = kb + j, k1 = kb + 16 + j;
            float v0 = p[j]      * scsh[k0] + scsh[128 + k0];
            float v1 = p[16 + j] * scsh[k1] + scsh[128 + k1];
            v0 = v0 > 0.f ? v0 : pa * v0;
            v1 = v1 > 0.f ? v1 : pa * v1;
            a[j]     = (_Float16)v0;
            a[8 + j] = (_Float16)v1;
        }
        v16h b = *(const v16h*)(Wp + ((size_t)kt * 32 + lane) * 16);
        acc = __builtin_amdgcn_wmma_f32_16x16x32_f16(
            false, a, false, b, (short)0, acc, false, false);
    }
    if (coln < NCLS) {
#pragma unroll
        for (int r = 0; r < 8; ++r)
            Out[(size_t)(row0 + mh + r) * NCLS + coln] = acc[r];
    }
}

// ---------------------------------------------------------------------------
// Graph / elementwise kernels
// ---------------------------------------------------------------------------
__global__ void fill_f32(float* __restrict__ p, float v, int n) {
    int i = blockIdx.x * blockDim.x + threadIdx.x;
    if (i < n) p[i] = v;
}

__global__ void deg_accum(const long long* __restrict__ ei, float* __restrict__ deg, int ne) {
    int j = blockIdx.x * blockDim.x + threadIdx.x;
    if (j < ne) atomAddF(&deg[(int)ei[(size_t)ne + j]], 1.0f);
}

__global__ void rsqrt_k(const float* __restrict__ deg, float* __restrict__ dinv, int n) {
    int i = blockIdx.x * blockDim.x + threadIdx.x;
    if (i < n) dinv[i] = rsqrtf(deg[i]);
}

__global__ void copy_k(const float* __restrict__ src, float* __restrict__ dst, int n) {
    int i = blockIdx.x * blockDim.x + threadIdx.x;
    if (i < n) dst[i] = src[i];
}

// h_new[i,f] = dinv[i]^2 * h[i,f]   (self-loop term; also initializes h_new)
__global__ void selfloop_init(const float* __restrict__ h, const float* __restrict__ dinv,
                              float* __restrict__ h_new, int n64) {
    int idx = blockIdx.x * blockDim.x + threadIdx.x;
    if (idx < n64) {
        float d = dinv[idx >> 6];
        h_new[idx] = d * d * h[idx];
    }
}

// One wave per edge: 64 features, 2 per lane; coalesced gather + fp32 atomics.
__global__ void edge_scatter(const float* __restrict__ h, const float* __restrict__ eh,
                             const float* __restrict__ dinv, const long long* __restrict__ ei,
                             float* __restrict__ h_new, int ne) {
    int wave = blockIdx.x * (blockDim.x >> 5) + (threadIdx.x >> 5);
    if (wave >= ne) return;
    int lane = threadIdx.x & 31;
    int s = (int)ei[wave];
    int d = (int)ei[(size_t)ne + wave];
    float nrm = dinv[s] * dinv[d];
    const float* hs = h  + (size_t)s    * 64;
    const float* ep = eh + (size_t)wave * 64;
    float*       hd = h_new + (size_t)d * 64;
    float m0 = nrm * (hs[lane]      + ep[lane]);
    float m1 = nrm * (hs[lane + 32] + ep[lane + 32]);
    atomAddF(&hd[lane],      m0);
    atomAddF(&hd[lane + 32], m1);
}

// h = leaky_relu(h_new, 0.2); h_sum += h
__global__ void act_accum(const float* __restrict__ h_new, float* __restrict__ h,
                          float* __restrict__ h_sum, int n64) {
    int idx = blockIdx.x * blockDim.x + threadIdx.x;
    if (idx < n64) {
        float v = h_new[idx];
        v = v > 0.f ? v : 0.2f * v;
        h[idx] = v;
        h_sum[idx] += v;
    }
}

// Per-column partial sums (sum, sumsq) with strided rows + atomics.
__global__ void bn_stats(const float* __restrict__ z, float* __restrict__ stats, int n) {
    int tid  = blockIdx.x * blockDim.x + threadIdx.x;
    int col  = tid & 127;
    int grp  = tid >> 7;
    int ngrp = (gridDim.x * blockDim.x) >> 7;
    float s = 0.f, s2 = 0.f;
    for (int r = grp; r < n; r += ngrp) {
        float v = z[(size_t)r * 128 + col];
        s += v; s2 += v * v;
    }
    atomAddF(&stats[col],       s);
    atomAddF(&stats[128 + col], s2);
}

// Fold BN into per-column scale/shift.
__global__ void bn_finalize(const float* __restrict__ stats, const float* __restrict__ gamma,
                            const float* __restrict__ beta, float* __restrict__ scsh, int n) {
    int c = threadIdx.x;
    if (c < 128) {
        float mean = stats[c] / (float)n;
        float var  = stats[128 + c] / (float)n - mean * mean;
        float sc   = gamma[c] * rsqrtf(var + BN_EPS);
        scsh[c]        = sc;
        scsh[128 + c]  = beta[c] - mean * sc;
    }
}

// ---------------------------------------------------------------------------
extern "C" void kernel_launch(void* const* d_in, const int* in_sizes, int n_in,
                              void* d_out, int out_size, void* d_ws, size_t ws_size,
                              hipStream_t stream) {
    const float* x        = (const float*)d_in[0];
    const float* e        = (const float*)d_in[1];
    const float* c        = (const float*)d_in[2];
    const float* w_node   = (const float*)d_in[3];
    const float* b_node   = (const float*)d_in[4];
    const float* w_edge   = (const float*)d_in[5];
    const float* b_edge   = (const float*)d_in[6];
    const float* w_out1   = (const float*)d_in[7];
    const float* b_out1   = (const float*)d_in[8];
    const float* bn_gamma = (const float*)d_in[9];
    const float* bn_beta  = (const float*)d_in[10];
    const float* prelu_a  = (const float*)d_in[11];
    const float* w_out2   = (const float*)d_in[12];
    const float* b_out2   = (const float*)d_in[13];
    const long long* ei   = (const long long*)d_in[14];
    float* out = (float*)d_out;

    char* base = (char*)d_ws;
    size_t off = 0;
    auto take = [&](size_t bytes) -> void* {
        size_t o = (off + 255) & ~(size_t)255;
        off = o + bytes;
        return (void*)(base + o);
    };
    _Float16* wnode_p = (_Float16*)take((size_t)4 * 4 * 32 * 16 * 2);
    _Float16* wedge_p = (_Float16*)take((size_t)1 * 4 * 32 * 16 * 2);
    _Float16* wout1_p = (_Float16*)take((size_t)3 * 8 * 32 * 16 * 2);
    _Float16* wout2_p = (_Float16*)take((size_t)4 * 1 * 32 * 16 * 2);
    float* h     = (float*)take((size_t)NN * 64 * 4);
    float* h_new = (float*)take((size_t)NN * 64 * 4);
    float* h_sum = (float*)take((size_t)NN * 64 * 4);
    float* eh    = (float*)take((size_t)NE * 64 * 4);
    float* deg   = (float*)take((size_t)NN * 4);
    float* dinv  = (float*)take((size_t)NN * 4);
    float* z     = (float*)take((size_t)NN * 128 * 4);
    float* stats = (float*)take(256 * 4);
    float* scsh  = (float*)take(256 * 4);

    // --- pack weights to f16 fragment layout ---
    pack_w<<<2, 256, 0, stream>>>(w_node, wnode_p, 128, 64, 64);
    pack_w<<<1, 128, 0, stream>>>(w_edge, wedge_p, 32, 64, 64);
    pack_w<<<3, 256, 0, stream>>>(w_out1, wout1_p, 96, 128, 128);
    pack_w<<<1, 128, 0, stream>>>(w_out2, wout2_p, 128, 10, 16);

    const int mt_n = NN / 16;   // 6250
    const int mt_e = NE / 16;   // 62500
    const int n64  = NN * 64;

    // --- input GEMMs (WMMA f16 -> f32) ---
    gemm_wmma<4, 4><<<(mt_n + 7) / 8, 256, 0, stream>>>(x, 128, wnode_p, b_node, h, 64, 64, mt_n);
    gemm_wmma<1, 4><<<(mt_e + 7) / 8, 256, 0, stream>>>(e, 32, wedge_p, b_edge, eh, 64, 64, mt_e);

    // --- degrees (with self-loops) and dinv = deg^-1/2 ---
    fill_f32<<<(NN + 255) / 256, 256, 0, stream>>>(deg, 1.0f, NN);
    deg_accum<<<(NE + 255) / 256, 256, 0, stream>>>(ei, deg, NE);
    rsqrt_k<<<(NN + 255) / 256, 256, 0, stream>>>(deg, dinv, NN);

    // h_sum = h
    copy_k<<<(n64 + 255) / 256, 256, 0, stream>>>(h, h_sum, n64);

    // --- 3 GCN layers ---
    for (int l = 0; l < 3; ++l) {
        selfloop_init<<<(n64 + 255) / 256, 256, 0, stream>>>(h, dinv, h_new, n64);
        edge_scatter<<<NE / 8, 256, 0, stream>>>(h, eh, dinv, ei, h_new, NE);
        act_accum<<<(n64 + 255) / 256, 256, 0, stream>>>(h_new, h, h_sum, n64);
    }

    // --- output head ---
    gemm_out1<<<(mt_n + 7) / 8, 256, 0, stream>>>(h_sum, c, wout1_p, b_out1, z, mt_n);
    fill_f32<<<1, 256, 0, stream>>>(stats, 0.f, 256);
    bn_stats<<<512, 256, 0, stream>>>(z, stats, NN);
    bn_finalize<<<1, 128, 0, stream>>>(stats, bn_gamma, bn_beta, scsh, NN);
    gemm_out2<<<(mt_n + 7) / 8, 256, 0, stream>>>(z, scsh, prelu_a, wout2_p, b_out2, out, mt_n);

    (void)in_sizes; (void)n_in; (void)out_size; (void)ws_size;
}
